// DKEF_90340342104307
// MI455X (gfx1250) — compile-verified, matrix operand order checked by reading, not run on hardware
//
#include <hip/hip_runtime.h>
#include <hip/hip_bf16.h>

typedef __attribute__((ext_vector_type(16))) _Float16 v16h;
typedef __attribute__((ext_vector_type(8)))  _Float16 v8h;
typedef __attribute__((ext_vector_type(8)))  float    v8f;

#define KK 3     // kernels
#define DD 128   // input dim
#define HH 64    // hidden dim
#define BN 64    // output tile rows (x)
#define BM 64    // output tile cols (z)
#define HP 72    // padded LDS row (H + 8 halfs) -> 144B stride, conflict-free b128 frag loads

// ---------------------------------------------------------------------------
// Phase 1: phi(x) = W3 @ softplus(W2 @ softplus(W1 @ x + b1) + b2) + b3
// One block = (row, k). 64 threads, one per hidden unit. f32 math, then
// store f16 feature row (for WMMA) + squared norm (for distance expansion).
// ---------------------------------------------------------------------------
__device__ __forceinline__ float softplus_f(float v) {
    return fmaxf(v, 0.f) + log1pf(__expf(-fabsf(v)));
}

__global__ __launch_bounds__(64) void dkef_phi(
    const float* __restrict__ X,
    const float* __restrict__ W1, const float* __restrict__ b1,
    const float* __restrict__ W2, const float* __restrict__ b2,
    const float* __restrict__ W3, const float* __restrict__ b3,
    _Float16* __restrict__ Fh, float* __restrict__ Fn, int rows)
{
    const int row = blockIdx.x;
    const int k   = blockIdx.y;
    const int t   = threadIdx.x;   // 0..63

    __shared__ float xs[DD];
    __shared__ float h1[HH];
    __shared__ float h2[HH];
    __shared__ float red[HH];

    xs[t]      = X[(size_t)row * DD + t];
    xs[t + 64] = X[(size_t)row * DD + t + 64];
    __syncthreads();

    const float* w1r = W1 + ((size_t)k * HH + t) * DD;
    float acc = b1[k * HH + t];
    #pragma unroll 8
    for (int d = 0; d < DD; ++d) acc = fmaf(w1r[d], xs[d], acc);
    h1[t] = softplus_f(acc);
    __syncthreads();

    const float* w2r = W2 + ((size_t)k * HH + t) * HH;
    acc = b2[k * HH + t];
    #pragma unroll 8
    for (int h = 0; h < HH; ++h) acc = fmaf(w2r[h], h1[h], acc);
    h2[t] = softplus_f(acc);
    __syncthreads();

    const float* w3r = W3 + ((size_t)k * HH + t) * HH;
    acc = b3[k * HH + t];
    #pragma unroll 8
    for (int h = 0; h < HH; ++h) acc = fmaf(w3r[h], h2[h], acc);

    Fh[((size_t)k * rows + row) * HH + t] = (_Float16)acc;
    red[t] = acc * acc;
    __syncthreads();
    #pragma unroll
    for (int s = 32; s > 0; s >>= 1) {
        if (t < s) red[t] += red[t + s];
        __syncthreads();
    }
    if (t == 0) Fn[(size_t)k * rows + row] = red[0];
}

// ---------------------------------------------------------------------------
// Phase 2: fused pairwise kernel.
//   S[k] = fx_tile[k] @ fz_tile[k]^T  via v_wmma_f32_16x16x32_f16
//   out  = sum_k w[k] * exp(-inv2sig[k] * max(nx + nz - 2 S, 0))
// Block = 64x64 output tile, 256 threads = 8 wave32s; each wave owns a
// 16x32 slab -> 2 (16x16) WMMA tiles, each needing 2 chained WMMAs (H=64).
// ---------------------------------------------------------------------------
// Fragment load: per ISA 7.12.2, 16-bit A 16x32: lane L holds row M=L%16,
// VGPRs 0..3 hold K = (L/16)*8 + 0..7 (contig), VGPRs 4..7 hold K = 16 +
// (L/16)*8 + 0..7 -> two 16B LDS loads. B (= fz^T) is loaded with the
// mirrored addressing (lane%16 = output column).
__device__ __forceinline__ v16h ld_frag(const _Float16* rowp, int off) {
    v8h lo = *(const v8h*)(rowp + off);        // K = off .. off+7
    v8h hi = *(const v8h*)(rowp + off + 16);   // K = off+16 .. off+23
    v16h r;
    #pragma unroll
    for (int i = 0; i < 8; ++i) { r[i] = lo[i]; r[i + 8] = hi[i]; }
    return r;
}

__device__ __forceinline__ v8f wmma_f16(v16h a, v16h b, v8f c) {
    return __builtin_amdgcn_wmma_f32_16x16x32_f16(
        /*neg_a=*/false, a, /*neg_b=*/false, b,
        /*c_mod=*/(short)0, c, /*reuse_a=*/false, /*reuse_b=*/false);
}

__global__ __launch_bounds__(256) void dkef_pair(
    const _Float16* __restrict__ Fxh, const _Float16* __restrict__ Fzh,
    const float* __restrict__ Fxn, const float* __restrict__ Fzn,
    const float* __restrict__ log_sigma, const float* __restrict__ kw,
    float* __restrict__ out, int Nn, int Mm)
{
    __shared__ __align__(16) _Float16 sA[KK][BN][HP];   // 27648 B
    __shared__ __align__(16) _Float16 sB[KK][BM][HP];   // 27648 B
    __shared__ float sNx[KK][BN];
    __shared__ float sNz[KK][BM];

    const int tid = threadIdx.x;
    const int m0  = blockIdx.x * BM;
    const int n0  = blockIdx.y * BN;

    // ---- stage tiles to LDS (rows are 128B contiguous in global) ----
    for (int i = tid; i < KK * BN * 8; i += 256) {      // 1536 x uint4
        const int k = i / (BN * 8), rr = (i / 8) % BN, c = i % 8;
        *((uint4*)&sA[k][rr][0] + c) =
            *((const uint4*)(Fxh + ((size_t)k * Nn + n0 + rr) * HH) + c);
    }
    for (int i = tid; i < KK * BM * 8; i += 256) {
        const int k = i / (BM * 8), rr = (i / 8) % BM, c = i % 8;
        *((uint4*)&sB[k][rr][0] + c) =
            *((const uint4*)(Fzh + ((size_t)k * Mm + m0 + rr) * HH) + c);
    }
    for (int i = tid; i < KK * BN; i += 256)
        sNx[i / BN][i % BN] = Fxn[(size_t)(i / BN) * Nn + n0 + (i % BN)];
    for (int i = tid; i < KK * BM; i += 256)
        sNz[i / BM][i % BM] = Fzn[(size_t)(i / BM) * Mm + m0 + (i % BM)];
    __syncthreads();

    // ---- uniform scalars: softmax(kernel_weights), 1/(2*10^log_sigma) ----
    const float c0 = kw[0], c1 = kw[1], c2 = kw[2];
    const float mx = fmaxf(c0, fmaxf(c1, c2));
    const float e0 = __expf(c0 - mx), e1 = __expf(c1 - mx), e2 = __expf(c2 - mx);
    const float inv = 1.f / (e0 + e1 + e2);
    const float wks[KK] = { e0 * inv, e1 * inv, e2 * inv };
    float iss[KK];
    #pragma unroll
    for (int k = 0; k < KK; ++k)
        iss[k] = 0.5f * __expf(-2.3025850929940457f * log_sigma[k]);

    // ---- wave tiling: 8 waves, each owns rows [waveN*16,+16) x cols [waveM*32,+32) ----
    const int wave  = tid >> 5;
    const int lane  = tid & 31;
    const int ln    = lane & 15;        // row (A) / col (B,C) within 16
    const int g     = lane >> 4;        // lane group for K split / C row offset
    const int waveN = wave >> 1;        // 0..3
    const int waveM = wave & 1;         // 0..1
    const int rowBase = waveN * 16;
    const int colBase = waveM * 32;

    float o[2][8];
    #pragma unroll
    for (int mt = 0; mt < 2; ++mt)
        #pragma unroll
        for (int r = 0; r < 8; ++r) o[mt][r] = 0.f;

    #pragma unroll
    for (int k = 0; k < KK; ++k) {
        const _Float16* aRow = &sA[k][rowBase + ln][0];
        const v16h a0 = ld_frag(aRow, g * 8);          // K =  0..31
        const v16h a1 = ld_frag(aRow, 32 + g * 8);     // K = 32..63
        float nxv[8];
        #pragma unroll
        for (int r = 0; r < 8; ++r) nxv[r] = sNx[k][rowBase + 8 * g + r];

        #pragma unroll
        for (int mt = 0; mt < 2; ++mt) {
            const _Float16* bRow = &sB[k][colBase + mt * 16 + ln][0];
            const v16h b0 = ld_frag(bRow, g * 8);
            const v16h b1 = ld_frag(bRow, 32 + g * 8);

            v8f acc = {};
            acc = wmma_f16(a0, b0, acc);
            acc = wmma_f16(a1, b1, acc);

            const float nzv = sNz[k][colBase + mt * 16 + ln];
            #pragma unroll
            for (int r = 0; r < 8; ++r) {
                const float sq = fmaxf(nxv[r] + nzv - 2.f * acc[r], 0.f);
                o[mt][r] += wks[k] * __expf(-iss[k] * sq);
            }
        }
    }

    // ---- streaming store: 256MB output -> non-temporal, keep L2 for fx/fz ----
    float* op = out + (size_t)(n0 + rowBase + 8 * g) * Mm + (m0 + colBase + ln);
    #pragma unroll
    for (int mt = 0; mt < 2; ++mt)
        #pragma unroll
        for (int r = 0; r < 8; ++r)
            __builtin_nontemporal_store(o[mt][r], op + (size_t)r * Mm + mt * 16);
}

// ---------------------------------------------------------------------------
// Host launcher
// ---------------------------------------------------------------------------
extern "C" void kernel_launch(void* const* d_in, const int* in_sizes, int n_in,
                              void* d_out, int out_size, void* d_ws, size_t ws_size,
                              hipStream_t stream)
{
    const float* x  = (const float*)d_in[0];
    const float* z  = (const float*)d_in[1];
    const float* W1 = (const float*)d_in[2];
    const float* b1 = (const float*)d_in[3];
    const float* W2 = (const float*)d_in[4];
    const float* b2 = (const float*)d_in[5];
    const float* W3 = (const float*)d_in[6];
    const float* b3 = (const float*)d_in[7];
    const float* ls = (const float*)d_in[8];
    const float* kw = (const float*)d_in[9];
    float* out = (float*)d_out;

    const int N_ = in_sizes[0] / DD;   // 16384
    const int M_ = in_sizes[1] / DD;   // 4096

    char* ws = (char*)d_ws;
    size_t off = 0;
    _Float16* Fxh = (_Float16*)(ws + off); off += (size_t)KK * N_ * HH * sizeof(_Float16);
    off = (off + 255) & ~(size_t)255;
    _Float16* Fzh = (_Float16*)(ws + off); off += (size_t)KK * M_ * HH * sizeof(_Float16);
    off = (off + 255) & ~(size_t)255;
    float* Fxn = (float*)(ws + off); off += (size_t)KK * N_ * sizeof(float);
    off = (off + 255) & ~(size_t)255;
    float* Fzn = (float*)(ws + off);

    dkef_phi<<<dim3(N_, KK), 64, 0, stream>>>(x, W1, b1, W2, b2, W3, b3, Fxh, Fxn, N_);
    dkef_phi<<<dim3(M_, KK), 64, 0, stream>>>(z, W1, b1, W2, b2, W3, b3, Fzh, Fzn, M_);
    dkef_pair<<<dim3(M_ / BM, N_ / BN), 256, 0, stream>>>(
        Fxh, Fzh, Fxn, Fzn, ls, kw, out, N_, M_);
}
